// APPNP_Net_67018669687299
// MI455X (gfx1250) — matrix operand order repaired and checked by reading, not compiled, not run on hardware
//
#include <hip/hip_runtime.h>
#include <hip/hip_bf16.h>

typedef __attribute__((ext_vector_type(16))) __bf16 v16bf;
typedef __attribute__((ext_vector_type(8)))  float  v8f;

#define ALPHA  0.1f
#define BETA   0.9f
#define KSTEPS 10
#define IN_DIM  256
#define HID_DIM 256
#define OUT_DIM 64

// ---------------------------------------------------------------------------
// Pre-swizzle a column-major-consumed weight matrix W[K x Ncol] (fp32) into
// WMMA B-fragment layout, bf16.  Fragment (jt, kc) covers cols jt*16..+15 and
// K rows kc*32..+31.  Lane layout (ISA 7.12.2, 16-bit B 32x16):
//   lanes 0-15: K = 0..7,16..23  (element h -> K = h<8 ? h : h + 8)
//   lanes 16-31: K = 8..15,24..31 (element h -> K = h<8 ? 8 + h : 16 + h)
//   column = jt*16 + (lane & 15)
// Stored so each lane's 16 halves (32 bytes) are contiguous.
// ---------------------------------------------------------------------------
__global__ void swizzle_w_kernel(const float* __restrict__ W,
                                 __bf16* __restrict__ WB,
                                 int Kdim, int Ncol) {
  int t = blockIdx.x * blockDim.x + threadIdx.x;
  int kcN = Kdim / 32;
  int total = (Ncol / 16) * kcN * 32 * 16;
  if (t >= total) return;
  int h    = t & 15;
  int lane = (t >> 4) & 31;
  int rest = t >> 9;
  int kc   = rest % kcN;
  int jt   = rest / kcN;
  int K    = kc * 32 + ((h >= 8) ? 16 : 0) + ((lane >= 16) ? 8 : 0) + (h & 7);
  int col  = jt * 16 + (lane & 15);
  WB[t] = (__bf16)W[(long long)K * Ncol + col];
}

// ---------------------------------------------------------------------------
// Fused MLP: h0 = relu(x @ W1 + b1) @ W2 + b2     (WMMA bf16 -> f32)
// One wave (32 threads) per 16-row tile.  A-fragment layout (16-bit A 16x32):
//   lane L, row = L & 15; element h -> K = (h>=8?16:0) + (L>=16?8:0) + (h&7)
// ---------------------------------------------------------------------------
__global__ __launch_bounds__(32) void mlp_wmma_kernel(
    const float* __restrict__ x,
    const float* __restrict__ b1,
    const float* __restrict__ b2,
    const __bf16* __restrict__ W1B,
    const __bf16* __restrict__ W2B,
    float* __restrict__ h0, int N) {
  __shared__ __bf16 h1s[16 * HID_DIM];  // relu(x@W1+b1) tile, bf16

  const int lane = threadIdx.x;
  const int mrow = lane & 15;
  const int hi   = lane >> 4;          // 0 or 1 (lane half)
  const int row0 = blockIdx.x * 16;

  int rowi = row0 + mrow;
  if (rowi > N - 1) rowi = N - 1;      // clamp (N may not be /16)
  const float* xrow = x + (long long)rowi * IN_DIM;

  // Preload all A fragments of the x tile (8 K-chunks) into registers.
  v16bf afr[8];
#pragma unroll
  for (int kc = 0; kc < 8; ++kc) {
    const int kb = kc * 32 + hi * 8;
    v16bf a;
#pragma unroll
    for (int i = 0; i < 8; ++i) a[i]     = (__bf16)xrow[kb + i];
#pragma unroll
    for (int i = 0; i < 8; ++i) a[8 + i] = (__bf16)xrow[kb + 16 + i];
    afr[kc] = a;
  }

  // ---- GEMM1: hidden = relu(x @ W1 + b1), 16 column tiles of 16 ----
#pragma unroll 1
  for (int jt = 0; jt < 16; ++jt) {
    v8f acc = {};
#pragma unroll
    for (int kc = 0; kc < 8; ++kc) {
      v16bf b = *(const v16bf*)(W1B + (((jt * 8 + kc) * 32 + lane) << 4));
      acc = __builtin_amdgcn_wmma_f32_16x16x32_bf16(
          false, afr[kc], false, b, (short)0, acc, false, false);
    }
    const int col = jt * 16 + mrow;       // C layout: N = lane & 15
    const float bias = b1[col];
#pragma unroll
    for (int r = 0; r < 8; ++r) {         // C layout: M = r + hi*8
      float v = acc[r] + bias;
      v = v > 0.0f ? v : 0.0f;
      h1s[(r + hi * 8) * HID_DIM + col] = (__bf16)v;
    }
  }
  __syncthreads();  // single-wave WG: ordering only

  // ---- GEMM2: h0 = hidden @ W2 + b2, 4 column tiles of 16 ----
  const __bf16* hrow = h1s + mrow * HID_DIM;
#pragma unroll 1
  for (int ot = 0; ot < 4; ++ot) {
    v8f acc = {};
#pragma unroll
    for (int kc = 0; kc < 8; ++kc) {
      const int kb = kc * 32 + hi * 8;
      v16bf a;
#pragma unroll
      for (int i = 0; i < 8; ++i) a[i]     = hrow[kb + i];
#pragma unroll
      for (int i = 0; i < 8; ++i) a[8 + i] = hrow[kb + 16 + i];
      v16bf b = *(const v16bf*)(W2B + (((ot * 8 + kc) * 32 + lane) << 4));
      acc = __builtin_amdgcn_wmma_f32_16x16x32_bf16(
          false, a, false, b, (short)0, acc, false, false);
    }
    const int col = ot * 16 + mrow;
    const float bias = b2[col];
#pragma unroll
    for (int r = 0; r < 8; ++r) {
      const int rl = r + hi * 8;
      if (row0 + rl < N)
        h0[(long long)(row0 + rl) * OUT_DIM + col] = acc[r] + bias;
    }
  }
}

// ---------------------------------------------------------------------------
// Degree / normalization kernels
// ---------------------------------------------------------------------------
__global__ void deg_init_kernel(float* __restrict__ deg, int N) {
  int i = blockIdx.x * blockDim.x + threadIdx.x;
  if (i < N) deg[i] = 1.0f;  // self-loop
}

__global__ void deg_acc_kernel(float* __restrict__ deg,
                               const long long* __restrict__ ei, long long E) {
  long long e = blockIdx.x * (long long)blockDim.x + threadIdx.x;
  if (e < E) atomicAdd(&deg[ei[E + e]], 1.0f);  // dst = ei[1][e]
}

__global__ void dinv_kernel(float* __restrict__ deg, int N) {
  int i = blockIdx.x * blockDim.x + threadIdx.x;
  if (i < N) {
    float d = deg[i];
    deg[i] = d > 0.0f ? rsqrtf(d) : 0.0f;  // in-place deg -> dinv
  }
}

__global__ void norm_kernel(float* __restrict__ norm,
                            const float* __restrict__ dinv,
                            const long long* __restrict__ ei, long long E) {
  long long e = blockIdx.x * (long long)blockDim.x + threadIdx.x;
  if (e < E) norm[e] = dinv[ei[e]] * dinv[ei[E + e]];
}

// ---------------------------------------------------------------------------
// Propagation step.  Node part initializes hnext with teleport + self-loop:
//   hnext = ALPHA*h0 + BETA*dinv[i]^2*h        (writes every element)
// Edge part scatter-adds BETA*norm[e]*h[src] into hnext[dst].
// ---------------------------------------------------------------------------
__global__ void node_prop_kernel(float* __restrict__ hnext,
                                 const float* __restrict__ h0,
                                 const float* __restrict__ h,
                                 const float* __restrict__ dinv, int N) {
  long long t = blockIdx.x * (long long)blockDim.x + threadIdx.x;
  if (t >= (long long)N * OUT_DIM) return;
  int i = (int)(t >> 6);
  float di = dinv[i];
  hnext[t] = ALPHA * h0[t] + BETA * di * di * h[t];
}

__global__ void edge_prop_kernel(float* __restrict__ hnext,
                                 const float* __restrict__ h,
                                 const float* __restrict__ norm,
                                 const long long* __restrict__ ei, long long E) {
  long long wid = blockIdx.x * (long long)(blockDim.x >> 5) + (threadIdx.x >> 5);
  int lane = threadIdx.x & 31;
  if (wid >= E) return;
  long long s = ei[wid];
  long long d = ei[E + wid];
  float w = BETA * norm[wid];
  const float2 v = *(const float2*)(h + s * OUT_DIM + lane * 2);
  float* dst = hnext + d * OUT_DIM + lane * 2;
  atomicAdd(dst + 0, w * v.x);
  atomicAdd(dst + 1, w * v.y);
}

// ---------------------------------------------------------------------------
// log_softmax over 64 features: one wave per node, 2 feats per lane.
// ---------------------------------------------------------------------------
__global__ void logsoftmax_kernel(float* __restrict__ out,
                                  const float* __restrict__ h, int N) {
  int wid = blockIdx.x * (blockDim.x >> 5) + (threadIdx.x >> 5);
  int lane = threadIdx.x & 31;
  if (wid >= N) return;
  const float2 v = *(const float2*)(h + (long long)wid * OUT_DIM + lane * 2);
  float m = fmaxf(v.x, v.y);
#pragma unroll
  for (int o = 16; o > 0; o >>= 1) m = fmaxf(m, __shfl_xor(m, o, 32));
  float s = expf(v.x - m) + expf(v.y - m);
#pragma unroll
  for (int o = 16; o > 0; o >>= 1) s += __shfl_xor(s, o, 32);
  float ls = logf(s);
  float2 r;
  r.x = v.x - m - ls;
  r.y = v.y - m - ls;
  *(float2*)(out + (long long)wid * OUT_DIM + lane * 2) = r;
}

// ---------------------------------------------------------------------------
extern "C" void kernel_launch(void* const* d_in, const int* in_sizes, int n_in,
                              void* d_out, int out_size, void* d_ws, size_t ws_size,
                              hipStream_t stream) {
  const float*     x  = (const float*)d_in[0];
  const long long* ei = (const long long*)d_in[1];
  const float*     W1 = (const float*)d_in[2];
  const float*     b1 = (const float*)d_in[3];
  const float*     W2 = (const float*)d_in[4];
  const float*     b2 = (const float*)d_in[5];
  float* outp = (float*)d_out;

  const int       N = in_sizes[0] / IN_DIM;
  const long long E = (long long)in_sizes[1] / 2;

  // ---- workspace carve-up ----
  char* ws = (char*)d_ws;
  float* h0 = (float*)ws;   ws += (size_t)N * OUT_DIM * sizeof(float);
  float* hA = (float*)ws;   ws += (size_t)N * OUT_DIM * sizeof(float);
  float* hB = (float*)ws;   ws += (size_t)N * OUT_DIM * sizeof(float);
  float* dinv = (float*)ws; ws += (size_t)N * sizeof(float);
  float* norm = (float*)ws; ws += (size_t)E * sizeof(float);
  __bf16* W1B = (__bf16*)ws; ws += (size_t)HID_DIM * IN_DIM * sizeof(__bf16);
  __bf16* W2B = (__bf16*)ws; ws += (size_t)OUT_DIM * HID_DIM * sizeof(__bf16);
  (void)ws_size; (void)n_in; (void)out_size;

  // ---- weight pre-swizzle into WMMA B-fragment layout (bf16) ----
  {
    int tot1 = (HID_DIM / 16) * (IN_DIM / 32) * 32 * 16;   // 65536
    swizzle_w_kernel<<<(tot1 + 255) / 256, 256, 0, stream>>>(W1, W1B, IN_DIM, HID_DIM);
    int tot2 = (OUT_DIM / 16) * (HID_DIM / 32) * 32 * 16;  // 16384
    swizzle_w_kernel<<<(tot2 + 255) / 256, 256, 0, stream>>>(W2, W2B, HID_DIM, OUT_DIM);
  }

  // ---- fused MLP via WMMA ----
  mlp_wmma_kernel<<<(N + 15) / 16, 32, 0, stream>>>(x, b1, b2, W1B, W2B, h0, N);

  // ---- GCN normalization (self-loops included) ----
  deg_init_kernel<<<(N + 255) / 256, 256, 0, stream>>>(dinv, N);
  deg_acc_kernel<<<(int)((E + 255) / 256), 256, 0, stream>>>(dinv, ei, E);
  dinv_kernel<<<(N + 255) / 256, 256, 0, stream>>>(dinv, N);
  norm_kernel<<<(int)((E + 255) / 256), 256, 0, stream>>>(norm, dinv, ei, E);

  // ---- K-step APPNP propagation (L2-resident working set) ----
  const float* hcur = h0;
  for (int k = 0; k < KSTEPS; ++k) {
    float* hnext = (k & 1) ? hB : hA;
    long long nt = (long long)N * OUT_DIM;
    node_prop_kernel<<<(int)((nt + 255) / 256), 256, 0, stream>>>(hnext, h0, hcur, dinv, N);
    edge_prop_kernel<<<(int)((E + 7) / 8), 256, 0, stream>>>(hnext, hcur, norm, ei, E);
    hcur = hnext;
  }

  // ---- log_softmax ----
  logsoftmax_kernel<<<(N + 7) / 8, 256, 0, stream>>>(outp, hcur, N);
}
